// Recurrent_64226940944749
// MI455X (gfx1250) — compile-verified, matrix operand order checked by reading, not run on hardware
//
#include <hip/hip_runtime.h>

#define TT 512
#define BB 64
#define DD 512
#define HH 1024
#define H3 3072

typedef __bf16 bf16;
typedef __attribute__((ext_vector_type(16))) __bf16 bf16x16;
typedef __attribute__((ext_vector_type(8)))  float  floatx8;

union Frag { bf16x16 v; uint4 q[2]; };

__device__ __forceinline__ floatx8 wmma_bf16(bf16x16 a, bf16x16 b, floatx8 c) {
  // D = A(16x32 bf16) * B(32x16 bf16) + C(16x16 f32)
  return __builtin_amdgcn_wmma_f32_16x16x32_bf16(false, a, false, b, (short)0, c,
                                                 false, false);
}

// Direct global -> LDS async copy (ASYNCcnt tracked); bypasses the VGPR file.
__device__ __forceinline__ void async_ld_b128(const void* lds_dst, const void* gsrc) {
  unsigned l = (unsigned)(unsigned long long)lds_dst;  // low 32b = LDS byte addr
  asm volatile("global_load_async_to_lds_b128 %0, %1, off"
               :: "v"(l), "v"(gsrc) : "memory");
}
#define WAIT_ASYNC0() asm volatile("s_wait_asynccnt 0x0" ::: "memory")

__device__ __forceinline__ float sigm(float x) { return 1.0f / (1.0f + __expf(-x)); }
__device__ __forceinline__ float tanh_fast(float x) {
  float e = __expf(-2.0f * x);
  return (1.0f - e) / (1.0f + e);
}

// ---------------------------------------------------------------------------
__global__ void f32_to_bf16_kernel(const float* __restrict__ in,
                                   bf16* __restrict__ out, int n) {
  int i = blockIdx.x * blockDim.x + threadIdx.x;
  if (i < n) out[i] = (bf16)in[i];
}

__global__ void init_state(bf16* __restrict__ h0, unsigned* __restrict__ cnt) {
  int i = blockIdx.x * blockDim.x + threadIdx.x;
  if (i < BB * HH) h0[i] = (bf16)0.0f;
  if (i == 0) *cnt = 0u;
}

// ---------------------------------------------------------------------------
// gi[m, 0:3072] = A[m, :] @ W^T + bias   (M = 32768, N = 3072)
// Block tile 128x64, 8 waves as 4(rowpair) x 2(colhalf); each wave: 32x32 C
// (4 WMMA accumulators). Double-buffered LDS filled by async global->LDS.
// mode 0: A = Xbf [B,T,D], m -> (t,b), optional time reversal, K = 512.
// mode 1: A = ys0 row-major [m, 1024], K = 1024.
// ---------------------------------------------------------------------------
__global__ __launch_bounds__(256) void gemm_in_proj(
    const bf16* __restrict__ A, const bf16* __restrict__ W,
    const float* __restrict__ bias, float* __restrict__ gi,
    int K, int mode, int rev) {
  __shared__ bf16 As[2][128 * 32];
  __shared__ bf16 Bs[2][64 * 32];
  const int tid  = threadIdx.x;
  const int wave = tid >> 5;
  const int lane = tid & 31;
  const int lm   = lane & 15;
  const int hi   = (lane >> 4) & 1;
  const int rp   = wave & 3;          // 32-row pair
  const int ch   = wave >> 2;         // 32-col half
  const int m0   = blockIdx.x * 128;
  const int n0   = blockIdx.y * 64;

  auto stage = [&](int buf, int k0) {
#pragma unroll
    for (int u = 0; u < 2; ++u) {     // A: 128 rows x 4 chunks = 2 xfers/thread
      int i = tid * 2 + u;
      int r = i >> 2, c = i & 3;
      int m = m0 + r;
      const bf16* src;
      if (mode == 0) {
        int t = m >> 6, b = m & 63;
        int tt = rev ? (TT - 1 - t) : t;
        src = A + ((size_t)b * TT + tt) * DD + k0 + c * 8;
      } else {
        src = A + (size_t)m * HH + k0 + c * 8;
      }
      async_ld_b128(&As[buf][r * 32 + c * 8], src);
    }
    {                                  // B: 64 rows x 4 chunks = 1 xfer/thread
      int r = tid >> 2, c = tid & 3;
      async_ld_b128(&Bs[buf][r * 32 + c * 8],
                    W + (size_t)(n0 + r) * K + k0 + c * 8);
    }
  };

  floatx8 c00 = {}, c01 = {}, c10 = {}, c11 = {};
  const int nk = K >> 5;
  stage(0, 0);
  for (int kk = 0; kk < nk; ++kk) {
    const int cur = kk & 1;
    WAIT_ASYNC0();          // own async writes into buf[cur] done
    __syncthreads();        // everyone's writes visible; prev reads of nxt done
    if (kk + 1 < nk) stage(cur ^ 1, (kk + 1) << 5);
    Frag a0, a1, b0, b1;
    const int ar0 = rp * 32 + lm, ar1 = ar0 + 16;
    a0.q[0] = *(const uint4*)&As[cur][ar0 * 32 + hi * 8];
    a0.q[1] = *(const uint4*)&As[cur][ar0 * 32 + 16 + hi * 8];
    a1.q[0] = *(const uint4*)&As[cur][ar1 * 32 + hi * 8];
    a1.q[1] = *(const uint4*)&As[cur][ar1 * 32 + 16 + hi * 8];
    const int br0 = ch * 32 + lm, br1 = br0 + 16;
    b0.q[0] = *(const uint4*)&Bs[cur][br0 * 32 + hi * 8];
    b0.q[1] = *(const uint4*)&Bs[cur][br0 * 32 + 16 + hi * 8];
    b1.q[0] = *(const uint4*)&Bs[cur][br1 * 32 + hi * 8];
    b1.q[1] = *(const uint4*)&Bs[cur][br1 * 32 + 16 + hi * 8];
    c00 = wmma_bf16(a0.v, b0.v, c00);
    c01 = wmma_bf16(a0.v, b1.v, c01);
    c10 = wmma_bf16(a1.v, b0.v, c10);
    c11 = wmma_bf16(a1.v, b1.v, c11);
  }
  const int mr0 = m0 + rp * 32 + hi * 8;
  const int mr1 = mr0 + 16;
  const int nc0 = n0 + ch * 32 + lm;
  const int nc1 = nc0 + 16;
  const float bb0 = bias[nc0];
  const float bb1 = bias[nc1];
#pragma unroll
  for (int r = 0; r < 8; ++r) {
    gi[(size_t)(mr0 + r) * H3 + nc0] = c00[r] + bb0;
    gi[(size_t)(mr0 + r) * H3 + nc1] = c01[r] + bb1;
    gi[(size_t)(mr1 + r) * H3 + nc0] = c10[r] + bb0;
    gi[(size_t)(mr1 + r) * H3 + nc1] = c11[r] + bb1;
  }
}

// ---------------------------------------------------------------------------
// Persistent GRU recurrence: 64 blocks, block owns 16 h-columns. Per step t,
// gh(:, {j, j+H, j+2H}) = h @ Whh^T via WMMA with async double-buffered h in
// LDS and Whh streamed from L2; fp32 gates; grid-wide barrier per step.
// ---------------------------------------------------------------------------
__global__ __launch_bounds__(256) void gru_recurrence(
    const float* __restrict__ gi, const bf16* __restrict__ Whh,
    const float* __restrict__ bhh, bf16* __restrict__ hbuf,   // [2][64*1024]
    bf16* __restrict__ ys_bf, float* __restrict__ xh_out,
    float* __restrict__ hh_out, unsigned* __restrict__ counter,
    int mode, int dirOff) {
  __shared__ bf16  Hs[2][64 * 32];
  __shared__ float Gh[3][64 * 16];
  const int tid  = threadIdx.x;
  const int wave = tid >> 5;
  const int lane = tid & 31;
  const int lm   = lane & 15;
  const int hi   = (lane >> 4) & 1;
  const int j0   = blockIdx.x * 16;
  const int rt   = (wave < 4) ? wave : (wave - 4);

  for (int t = 0; t < TT; ++t) {
    const bf16* hcur = hbuf + (size_t)(t & 1) * (BB * HH);
    bf16*       hnxt = hbuf + (size_t)((t + 1) & 1) * (BB * HH);
    auto stageH = [&](int buf, int k0) {
      int r = tid >> 2, c = tid & 3;
      async_ld_b128(&Hs[buf][r * 32 + c * 8],
                    hcur + (size_t)r * HH + k0 + c * 8);
    };
    floatx8 c0 = {}, c1 = {};
    stageH(0, 0);
    for (int kk = 0; kk < (HH >> 5); ++kk) {
      const int cur = kk & 1;
      const int k0  = kk << 5;
      WAIT_ASYNC0();
      __syncthreads();
      if (kk + 1 < (HH >> 5)) stageH(cur ^ 1, k0 + 32);
      Frag a;
      const int arow = rt * 16 + lm;
      a.q[0] = *(const uint4*)&Hs[cur][arow * 32 + hi * 8];
      a.q[1] = *(const uint4*)&Hs[cur][arow * 32 + 16 + hi * 8];
      if (wave < 4) {
        Frag b0, b1;
        const bf16* w0 = Whh + (size_t)(j0 + lm) * HH + k0;
        const bf16* w1 = Whh + (size_t)(HH + j0 + lm) * HH + k0;
        __builtin_prefetch(w0 + 32, 0, 0);   // next K chunk (L2-resident)
        b0.q[0] = *(const uint4*)(w0 + hi * 8);
        b0.q[1] = *(const uint4*)(w0 + 16 + hi * 8);
        b1.q[0] = *(const uint4*)(w1 + hi * 8);
        b1.q[1] = *(const uint4*)(w1 + 16 + hi * 8);
        c0 = wmma_bf16(a.v, b0.v, c0);
        c1 = wmma_bf16(a.v, b1.v, c1);
      } else {
        Frag b2;
        const bf16* w2 = Whh + (size_t)(2 * HH + j0 + lm) * HH + k0;
        b2.q[0] = *(const uint4*)(w2 + hi * 8);
        b2.q[1] = *(const uint4*)(w2 + 16 + hi * 8);
        c0 = wmma_bf16(a.v, b2.v, c0);
      }
    }
    // exchange gate partials through LDS
#pragma unroll
    for (int r = 0; r < 8; ++r) {
      int bl = rt * 16 + r + hi * 8;
      if (wave < 4) {
        Gh[0][bl * 16 + lm] = c0[r];
        Gh[1][bl * 16 + lm] = c1[r];
      } else {
        Gh[2][bl * 16 + lm] = c0[r];
      }
    }
    __syncthreads();
    // fp32 gate math: 1024 (b, j) elements, 4 per thread
#pragma unroll
    for (int i = 0; i < 4; ++i) {
      int e = tid * 4 + i;
      int b = e >> 4, j = e & 15;
      size_t gbase = (size_t)(t * BB + b) * H3 + (j0 + j);
      float gir = gi[gbase];
      float giz = gi[gbase + HH];
      float gin = gi[gbase + 2 * HH];
      float ghr = Gh[0][b * 16 + j] + bhh[j0 + j];
      float ghz = Gh[1][b * 16 + j] + bhh[HH + j0 + j];
      float ghn = Gh[2][b * 16 + j] + bhh[2 * HH + j0 + j];
      float r = sigm(gir + ghr);
      float z = sigm(giz + ghz);
      float n = tanh_fast(gin + r * ghn);
      float hold = (float)hcur[(size_t)b * HH + (j0 + j)];
      float hnew = (1.0f - z) * n + z * hold;
      hnxt[(size_t)b * HH + (j0 + j)] = (bf16)hnew;
      if (mode == 0) {
        ys_bf[(size_t)(t * BB + b) * HH + (j0 + j)] = (bf16)hnew;
      } else {
        xh_out[(size_t)b * (TT * 2 * HH) + (size_t)t * (2 * HH) + dirOff + j0 + j] = hnew;
        if (t == TT - 1)
          hh_out[(size_t)b * (2 * HH) + dirOff + j0 + j] = hnew;
      }
    }
    // grid-wide barrier (release: fence+atomic; acquire: spin+fence)
    __syncthreads();
    if (tid == 0) {
      __threadfence();
      atomicAdd(counter, 1u);
      unsigned target = (unsigned)(t + 1) * gridDim.x;
      volatile unsigned* vc = counter;
      while (*vc < target) { __builtin_amdgcn_s_sleep(2); }
      __threadfence();
    }
    __syncthreads();
  }
}

// ---------------------------------------------------------------------------
extern "C" void kernel_launch(void* const* d_in, const int* in_sizes, int n_in,
                              void* d_out, int out_size, void* d_ws, size_t ws_size,
                              hipStream_t stream) {
  const float* X    = (const float*)d_in[0];
  const float* Wih0 = (const float*)d_in[1];
  const float* Whh0 = (const float*)d_in[2];
  const float* bih0 = (const float*)d_in[3];
  const float* bhh0 = (const float*)d_in[4];
  const float* Wih1 = (const float*)d_in[5];
  const float* Whh1 = (const float*)d_in[6];
  const float* bih1 = (const float*)d_in[7];
  const float* bhh1 = (const float*)d_in[8];
  float* out = (float*)d_out;

  char* ws = (char*)d_ws;
  size_t off = 0;
  bf16* Xbf   = (bf16*)(ws + off); off += (size_t)BB * TT * DD * 2;
  bf16* W0bf  = (bf16*)(ws + off); off += (size_t)H3 * DD * 2;
  bf16* Wh0bf = (bf16*)(ws + off); off += (size_t)H3 * HH * 2;
  bf16* W1bf  = (bf16*)(ws + off); off += (size_t)H3 * HH * 2;
  bf16* Wh1bf = (bf16*)(ws + off); off += (size_t)H3 * HH * 2;
  float* gi   = (float*)(ws + off); off += (size_t)TT * BB * H3 * 4;
  bf16* ys0   = (bf16*)(ws + off); off += (size_t)TT * BB * HH * 2;
  bf16* hbuf  = (bf16*)(ws + off); off += (size_t)2 * BB * HH * 2;
  unsigned* cnts = (unsigned*)(ws + off); off += 64;

  f32_to_bf16_kernel<<<(BB * TT * DD + 255) / 256, 256, 0, stream>>>(X, Xbf, BB * TT * DD);
  f32_to_bf16_kernel<<<(H3 * DD + 255) / 256, 256, 0, stream>>>(Wih0, W0bf, H3 * DD);
  f32_to_bf16_kernel<<<(H3 * HH + 255) / 256, 256, 0, stream>>>(Whh0, Wh0bf, H3 * HH);
  f32_to_bf16_kernel<<<(H3 * HH + 255) / 256, 256, 0, stream>>>(Wih1, W1bf, H3 * HH);
  f32_to_bf16_kernel<<<(H3 * HH + 255) / 256, 256, 0, stream>>>(Whh1, Wh1bf, H3 * HH);

  float* xh = out;                            // [B, T, 2H]
  float* hh = out + (size_t)BB * TT * 2 * HH; // [B, 2H]

  const dim3 gemm_grid(TT * BB / 128, H3 / 64);
  for (int dir = 0; dir < 2; ++dir) {
    // layer 0
    gemm_in_proj<<<gemm_grid, 256, 0, stream>>>(Xbf, W0bf, bih0, gi, DD, 0, dir);
    init_state<<<(BB * HH + 255) / 256, 256, 0, stream>>>(hbuf, &cnts[dir * 2 + 0]);
    gru_recurrence<<<HH / 16, 256, 0, stream>>>(gi, Wh0bf, bhh0, hbuf, ys0, nullptr,
                                                nullptr, &cnts[dir * 2 + 0], 0, 0);
    // layer 1
    gemm_in_proj<<<gemm_grid, 256, 0, stream>>>(ys0, W1bf, bih1, gi, HH, 1, 0);
    init_state<<<(BB * HH + 255) / 256, 256, 0, stream>>>(hbuf, &cnts[dir * 2 + 1]);
    gru_recurrence<<<HH / 16, 256, 0, stream>>>(gi, Wh1bf, bhh1, hbuf, nullptr, xh,
                                                hh, &cnts[dir * 2 + 1], 1, dir * HH);
  }
  (void)in_sizes; (void)n_in; (void)out_size; (void)ws_size;
}
